// GKAN_47459388621632
// MI455X (gfx1250) — matrix operand order repaired and checked by reading, not compiled, not run on hardware
//
#include <hip/hip_runtime.h>

typedef __attribute__((ext_vector_type(16))) _Float16 v16h;
typedef __attribute__((ext_vector_type(8)))  float    v8f;

// Trivially-constructible 16B chunks (HIP's uint4/float4 have ctors -> not union-safe)
struct alignas(16) U4 { unsigned int a, b, c, d; };
struct alignas(16) F4 { float x, y, z, w; };

#define DCH   64          // feature dim
#define NCOEF 8           // grid(5) + order(3)
#define KTOT  576         // 64 silu + 64*8 spline features
#define FSTR  584         // padded LDS row stride (halfwords); 1168B row, 16B aligned,
                          // 36-DWORD bank offset/row -> 16 rows hit disjoint bank groups
#define NGRAPHS 512

// ---------------- Cox-de Boor cubic B-spline basis (8 funcs, uniform grid) ----
__device__ __forceinline__ void bspl8(float x, float* out8) {
    const float h = 0.4f;
    float g[12];
#pragma unroll
    for (int j = 0; j < 12; ++j) g[j] = (float)(j - 3) * h - 1.0f;
    float b[11];
#pragma unroll
    for (int j = 0; j < 11; ++j) b[j] = (x >= g[j] && x < g[j + 1]) ? 1.0f : 0.0f;
#pragma unroll
    for (int k = 1; k <= 3; ++k) {
#pragma unroll
        for (int j = 0; j < 10; ++j) {
            if (j < 11 - k) {
                float left  = (x - g[j]) / (g[j + k] - g[j]);
                float right = (g[j + k + 1] - x) / (g[j + k + 1] - g[j + 1]);
                b[j] = left * b[j] + right * b[j + 1];
            }
        }
    }
#pragma unroll
    for (int j = 0; j < 8; ++j) out8[j] = b[j];
}

// silu + spline features for one (row, i) pair, written into an LDS feature row
__device__ __forceinline__ void write_features(_Float16* featA, int row, int i, float y) {
    float s = y / (1.0f + __expf(-y));              // silu
    featA[row * FSTR + i] = (_Float16)s;
    float b8[8];
    bspl8(y, b8);
    union { _Float16 h[8]; U4 q; } u;
#pragma unroll
    for (int g = 0; g < 8; ++g) u.h[g] = (_Float16)b8[g];
    *(U4*)&featA[row * FSTR + DCH + i * 8] = u.q;   // 16B-aligned
}

// ---------------- generic zero ----------------------------------------------
__global__ void zero_kernel(float* __restrict__ p, int n) {
    for (int i = blockIdx.x * blockDim.x + threadIdx.x; i < n; i += gridDim.x * blockDim.x)
        p[i] = 0.0f;
}

// ---------------- build combined fp16 weight  W[o][0:64]=base, W[o][64+8i+g]=spline*scaler
__global__ void prep_w(const float* __restrict__ base_w, const float* __restrict__ spline_w,
                       const float* __restrict__ scaler, _Float16* __restrict__ Wc,
                       int out_real, int out_pad) {
    int idx = blockIdx.x * blockDim.x + threadIdx.x;
    int total = out_pad * KTOT;
    if (idx >= total) return;
    int o = idx / KTOT, c = idx % KTOT;
    float v = 0.0f;
    if (o < out_real) {
        if (c < DCH) {
            v = base_w[o * DCH + c];
        } else {
            int t = c - DCH;
            int i = t >> 3, g = t & 7;
            v = spline_w[(o * DCH + i) * NCOEF + g] * scaler[o * DCH + i];
        }
    }
    Wc[o * KTOT + c] = (_Float16)v;
}

// ---------------- edge scatter-add: agg[dst] += feat[src] --------------------
__global__ void scatter_add(const float* __restrict__ feat, const int* __restrict__ ei,
                            float* __restrict__ agg, int nedges) {
    int t = blockIdx.x * blockDim.x + threadIdx.x;
    int e = t >> 4;
    if (e >= nedges) return;
    int c4 = (t & 15) * 4;
    int s = ei[e];
    int d = ei[nedges + e];
    F4 v = *(const F4*)&feat[(size_t)s * DCH + c4];
    float* p = &agg[(size_t)d * DCH + c4];
    atomicAdd(p + 0, v.x);
    atomicAdd(p + 1, v.y);
    atomicAdd(p + 2, v.z);
    atomicAdd(p + 3, v.w);
}

// ---------------- global add pool: pooled[batch[v]] += h[v] ------------------
__global__ void pool_kernel(const float* __restrict__ h, const int* __restrict__ batch,
                            float* __restrict__ pooled, int nnodes) {
    int t = blockIdx.x * blockDim.x + threadIdx.x;
    int v = t >> 4;
    if (v >= nnodes) return;
    int c4 = (t & 15) * 4;
    int g = batch[v];
    F4 val = *(const F4*)&h[(size_t)v * DCH + c4];
    float* p = &pooled[(size_t)g * DCH + c4];
    atomicAdd(p + 0, val.x);
    atomicAdd(p + 1, val.y);
    atomicAdd(p + 2, val.z);
    atomicAdd(p + 3, val.w);
}

// =============================================================================
// Big fused KAN linear (node layers): 64 rows x 64 outs per block, 4 waves.
// Wave w owns the 32x32 quadrant (rh = w>>1, ch = w&1) -> four 16x16 C tiles.
// Per K-step: 2 A frags + 2 B frags (8x ds_read_b128) feed 4 independent WMMAs.
// =============================================================================
template <bool ADD_AGG>
__global__ void __launch_bounds__(128)
kan_kernel_big(const float* __restrict__ xin, const float* __restrict__ agg,
               const _Float16* __restrict__ Wc, float* __restrict__ out,
               int nrows, int out_ld) {
    __shared__ alignas(16) _Float16 featA[64 * FSTR];   // 74,752 B
    __shared__ alignas(16) _Float16 Wlds[DCH * FSTR];   // 74,752 B

    const int rows0 = blockIdx.x * 64;
    const int tid = threadIdx.x;

    // ---- stage combined weights into LDS (16B chunks, row-restrided) ----
    for (int c = tid; c < DCH * (KTOT / 8); c += 128) {
        int o = c / (KTOT / 8), k = c % (KTOT / 8);
        *(U4*)&Wlds[o * FSTR + k * 8] = *(const U4*)&Wc[o * KTOT + k * 8];
    }

    // ---- build 64x576 fp16 feature tile in LDS ----
    for (int p = tid; p < 64 * DCH; p += 128) {
        int row = p >> 6, i = p & 63;
        int node = rows0 + row;
        float y = 0.0f;
        if (node < nrows) {
            y = xin[(size_t)node * DCH + i];
            if (ADD_AGG) y += agg[(size_t)node * DCH + i];
        }
        write_features(featA, row, i, y);
    }
    __syncthreads();

    // ---- WMMA macro-tile: 32x32 per wave ----
    const int L  = tid & 31;
    const int w  = tid >> 5;
    const int rh = w >> 1;                 // row half: rows rh*32 .. rh*32+31
    const int ch = w & 1;                  // col half: outs ch*32 .. ch*32+31
    const bool hi = (L >= 16);
    const int m = hi ? (L - 16) : L;

    union Frag { U4 q[2]; v16h v; };
    v8f c00 = {}, c01 = {}, c10 = {}, c11 = {};
#pragma unroll
    for (int kb = 0; kb < KTOT / 32; ++kb) {
        Frag a0, a1, b0, b1;
        // A 16x32 fp16 (ISA 7.12.2): lo lanes K{0..7,16..23}, hi lanes K{8..15,24..31}
        int abase = (rh * 32 + m) * FSTR + kb * 32 + (hi ? 8 : 0);
        a0.q[0] = *(const U4*)&featA[abase];
        a0.q[1] = *(const U4*)&featA[abase + 16];
        a1.q[0] = *(const U4*)&featA[abase + 16 * FSTR];
        a1.q[1] = *(const U4*)&featA[abase + 16 * FSTR + 16];
        // B 32x16 fp16: lane = column N; lo lanes K{0..7,16..23}, hi lanes K{8..15,24..31}
        int bbase = (ch * 32 + m) * FSTR + kb * 32 + (hi ? 8 : 0);
        b0.q[0] = *(const U4*)&Wlds[bbase];
        b0.q[1] = *(const U4*)&Wlds[bbase + 16];
        b1.q[0] = *(const U4*)&Wlds[bbase + 16 * FSTR];
        b1.q[1] = *(const U4*)&Wlds[bbase + 16 * FSTR + 16];

        c00 = __builtin_amdgcn_wmma_f32_16x16x32_f16(false, a0.v, false, b0.v, (short)0, c00, false, false);
        c01 = __builtin_amdgcn_wmma_f32_16x16x32_f16(false, a0.v, false, b1.v, (short)0, c01, false, false);
        c10 = __builtin_amdgcn_wmma_f32_16x16x32_f16(false, a1.v, false, b0.v, (short)0, c10, false, false);
        c11 = __builtin_amdgcn_wmma_f32_16x16x32_f16(false, a1.v, false, b1.v, (short)0, c11, false, false);
    }

    // ---- store: C/D layout: VGPR r -> M = r + 8*hi, N = L mod 16 ----
    const int n = L & 15;
    const v8f* tiles[4] = { &c00, &c01, &c10, &c11 };
#pragma unroll
    for (int ti = 0; ti < 2; ++ti) {
#pragma unroll
        for (int tj = 0; tj < 2; ++tj) {
            const v8f& cc = *tiles[ti * 2 + tj];
            int col = ch * 32 + tj * 16 + n;
            int rbase = rows0 + rh * 32 + ti * 16 + (hi ? 8 : 0);
#pragma unroll
            for (int r = 0; r < 8; ++r) {
                int node = rbase + r;
                if (node < nrows) out[(size_t)node * out_ld + col] = cc[r];
            }
        }
    }
}

// =============================================================================
// Small fused KAN linear (head): 16 rows, 16 padded outputs, one wave.
// =============================================================================
__global__ void __launch_bounds__(32)
kan_kernel_head(const float* __restrict__ xin, const _Float16* __restrict__ Wc,
                float* __restrict__ out, int nrows, int out_cols, int out_ld) {
    __shared__ alignas(16) _Float16 featA[16 * FSTR];
    __shared__ alignas(16) _Float16 Wlds[16 * FSTR];

    const int rows0 = blockIdx.x * 16;
    const int tid = threadIdx.x;

    for (int c = tid; c < 16 * (KTOT / 8); c += 32) {
        int o = c / (KTOT / 8), k = c % (KTOT / 8);
        *(U4*)&Wlds[o * FSTR + k * 8] = *(const U4*)&Wc[o * KTOT + k * 8];
    }
    for (int p = tid; p < 16 * DCH; p += 32) {
        int row = p >> 6, i = p & 63;
        int node = rows0 + row;
        float y = (node < nrows) ? xin[(size_t)node * DCH + i] : 0.0f;
        write_features(featA, row, i, y);
    }
    __syncthreads();

    const int L = tid;
    const bool hi = (L >= 16);
    const int m = hi ? (L - 16) : L;

    union Frag { U4 q[2]; v16h v; };
    v8f c = {};
#pragma unroll
    for (int kb = 0; kb < KTOT / 32; ++kb) {
        Frag a, b;
        int abase = m * FSTR + kb * 32 + (hi ? 8 : 0);
        a.q[0] = *(const U4*)&featA[abase];
        a.q[1] = *(const U4*)&featA[abase + 16];
        int bbase = m * FSTR + kb * 32 + (hi ? 8 : 0);
        b.q[0] = *(const U4*)&Wlds[bbase];
        b.q[1] = *(const U4*)&Wlds[bbase + 16];
        c = __builtin_amdgcn_wmma_f32_16x16x32_f16(false, a.v, false, b.v, (short)0, c, false, false);
    }

    int ncol = L & 15;
    if (ncol < out_cols) {
#pragma unroll
        for (int r = 0; r < 8; ++r) {
            int node = rows0 + r + (hi ? 8 : 0);
            if (node < nrows) out[(size_t)node * out_ld + ncol] = c[r];
        }
    }
}

extern "C" void kernel_launch(void* const* d_in, const int* in_sizes, int n_in,
                              void* d_out, int out_size, void* d_ws, size_t ws_size,
                              hipStream_t stream) {
    const float* x        = (const float*)d_in[0];
    const int*   ei       = (const int*)d_in[1];
    const int*   batch    = (const int*)d_in[2];
    const float* base_w0  = (const float*)d_in[3];
    const float* spline_w0= (const float*)d_in[4];
    const float* scaler0  = (const float*)d_in[5];
    const float* base_w1  = (const float*)d_in[6];
    const float* spline_w1= (const float*)d_in[7];
    const float* scaler1  = (const float*)d_in[8];
    const float* base_wh  = (const float*)d_in[9];
    const float* spline_wh= (const float*)d_in[10];
    const float* scaler_h = (const float*)d_in[11];
    float* out = (float*)d_out;

    const int nnodes = in_sizes[0] / DCH;
    const int nedges = in_sizes[1] / 2;

    // workspace layout (all 16B-aligned offsets)
    char* ws = (char*)d_ws;
    size_t off = 0;
    float* agg    = (float*)(ws + off); off += (size_t)nnodes * DCH * 4;   // also reused as h1
    float* h0     = (float*)(ws + off); off += (size_t)nnodes * DCH * 4;
    float* pooled = (float*)(ws + off); off += (size_t)NGRAPHS * DCH * 4;
    _Float16* Wc0 = (_Float16*)(ws + off); off += (size_t)DCH * KTOT * 2;
    _Float16* Wc1 = (_Float16*)(ws + off); off += (size_t)DCH * KTOT * 2;
    _Float16* Wch = (_Float16*)(ws + off); off += (size_t)16 * KTOT * 2;

    // pre-scale / pack weights to fp16
    prep_w<<<(DCH * KTOT + 255) / 256, 256, 0, stream>>>(base_w0, spline_w0, scaler0, Wc0, DCH, DCH);
    prep_w<<<(DCH * KTOT + 255) / 256, 256, 0, stream>>>(base_w1, spline_w1, scaler1, Wc1, DCH, DCH);
    prep_w<<<(16 * KTOT + 255) / 256, 256, 0, stream>>>(base_wh, spline_wh, scaler_h, Wch, 10, 16);

    const int featN = nnodes * DCH;
    const int zgrid = 2048;
    const int egrid = (nedges * 16 + 255) / 256;
    const int kgrid = (nnodes + 63) / 64;

    // ---- GIN layer 0 ----
    zero_kernel<<<zgrid, 256, 0, stream>>>(agg, featN);
    scatter_add<<<egrid, 256, 0, stream>>>(x, ei, agg, nedges);
    kan_kernel_big<true><<<kgrid, 128, 0, stream>>>(x, agg, Wc0, h0, nnodes, DCH);

    // ---- GIN layer 1 (h1 aliases agg: each block reads its rows before writing them) ----
    zero_kernel<<<zgrid, 256, 0, stream>>>(agg, featN);
    scatter_add<<<egrid, 256, 0, stream>>>(h0, ei, agg, nedges);
    kan_kernel_big<true><<<kgrid, 128, 0, stream>>>(h0, agg, Wc1, agg, nnodes, DCH);

    // ---- global add pool + KAN head ----
    zero_kernel<<<(NGRAPHS * DCH + 255) / 256, 256, 0, stream>>>(pooled, NGRAPHS * DCH);
    pool_kernel<<<(nnodes * 16 + 255) / 256, 256, 0, stream>>>(agg, batch, pooled, nnodes);
    kan_kernel_head<<<(NGRAPHS + 15) / 16, 32, 0, stream>>>(pooled, Wch, out, NGRAPHS, 10, 10);
}
